// SelectModel_64965675320070
// MI455X (gfx1250) — matrix operand order repaired and checked by reading, not compiled, not run on hardware
//
#include <hip/hip_runtime.h>
#include <math.h>

typedef __attribute__((ext_vector_type(16))) _Float16 v16h;
typedef __attribute__((ext_vector_type(8)))  _Float16 v8h;
typedef __attribute__((ext_vector_type(8)))  float    v8f;

#define R_   512
#define M_   64
#define T_   128
#define D0_  100
#define DL_  100
#define D1_  1000
#define DE_  100
#define KH_  128      // padded K for h and embn (f16 rows)
#define KP_  1024     // padded K for pooled activations P (f16 rows)

// ---- workspace layout (bytes), all 256B aligned ----
#define WEFF_OFF   0u          // f32 [400][3]
#define BEFF_OFF   8192u       // f32 [400]
#define H16_OFF    16384u      // f16 [32768][128]  = 8,388,608 B
#define W1H_OFF    8404992u    // f16 [1000][128]   =   256,000 B
#define P16_OFF    8660992u    // f16 [512][1024]   = 1,048,576 B
#define EMB_OFF    9709568u    // f32 [512][100]    =   204,800 B
#define EMBN_OFF   9914368u    // f16 [512][128]    =   131,072 B
#define W2H_OFF    10045440u   // f16 [100][1024]   =   204,800 B
// total ~10.25 MB

// ---------- WMMA helpers (ISA 16-bit A/B VGPR layouts) ----------
// A 16x32 f16: lanes 0-15 hold K=[0..7],[16..23]; lanes 16-31 hold K=[8..15],[24..31]
__device__ inline v16h load_a16(const _Float16* rowk, int khalf) {
  v8h lo = *reinterpret_cast<const v8h*>(rowk + 8 * khalf);
  v8h hi = *reinterpret_cast<const v8h*>(rowk + 16 + 8 * khalf);
  v16h r;
#pragma unroll
  for (int i = 0; i < 8; ++i) { r[i] = lo[i]; r[i + 8] = hi[i]; }
  return r;
}
// B 32x16 f16: lane n (=lane&15) holds K=[0..7],[16..23] (lanes 0-15) or K=[8..15],[24..31] (lanes 16-31)
__device__ inline v16h load_b16(const _Float16* rowk, int khalf) {
  v8h lo = *reinterpret_cast<const v8h*>(rowk + 8 * khalf);
  v8h hi = *reinterpret_cast<const v8h*>(rowk + 16 + 8 * khalf);
  v16h r;
#pragma unroll
  for (int i = 0; i < 8; ++i) { r[i] = lo[i]; r[i + 8] = hi[i]; }
  return r;
}
__device__ inline v16h zero16() {
  v16h z;
#pragma unroll
  for (int i = 0; i < 16; ++i) z[i] = (_Float16)0.f;
  return z;
}
__device__ inline v8f wmma_step(v16h a, v16h b, v8f c) {
  return __builtin_amdgcn_wmma_f32_16x16x32_f16(false, a, false, b, (short)0, c,
                                                false, false);
}
__device__ inline float sigmoidf_(float x) { return 1.f / (1.f + expf(-x)); }

// ---------- 0) fold linear0 + LSTM input path into [400x3] + bias ----------
// W_eff = W_ih @ W0 ; b_eff = W_ih@b0 + b_ih + W_hh@h0 + b_hh
__global__ __launch_bounds__(256) void prep_eff(const float* __restrict__ W0,
                                                const float* __restrict__ b0,
                                                const float* __restrict__ Wih,
                                                const float* __restrict__ bih,
                                                const float* __restrict__ Whh,
                                                const float* __restrict__ bhh,
                                                const float* __restrict__ h0,
                                                float* __restrict__ weff,
                                                float* __restrict__ beff) {
  int j = blockIdx.x * blockDim.x + threadIdx.x;
  if (j >= 4 * DL_) return;
  float a0 = 0.f, a1 = 0.f, a2 = 0.f, ab = 0.f, ah = 0.f;
  for (int d = 0; d < D0_; ++d) {
    float w = Wih[j * D0_ + d];
    a0 += w * W0[d * 3 + 0];
    a1 += w * W0[d * 3 + 1];
    a2 += w * W0[d * 3 + 2];
    ab += w * b0[d];
  }
  for (int d = 0; d < DL_; ++d) ah += Whh[j * DL_ + d] * h0[d];
  weff[j * 3 + 0] = a0;
  weff[j * 3 + 1] = a1;
  weff[j * 3 + 2] = a2;
  beff[j] = ab + bih[j] + ah + bhh[j];
}

// ---------- 1) repack W1 [1000][100] f32 -> [1000][128] f16 (zero pad) ----------
__global__ __launch_bounds__(256) void conv_w1(const float* __restrict__ W1,
                                               _Float16* __restrict__ w1h) {
  int idx = blockIdx.x * blockDim.x + threadIdx.x;  // 1000*128
  int k = idx & (KH_ - 1);
  int n = idx >> 7;
  w1h[(size_t)n * KH_ + k] = (k < D0_) ? (_Float16)W1[(size_t)n * D0_ + k] : (_Float16)0.f;
}

// ---------- 2) repack W2 [100][1000] f32 -> [100][1024] f16 (zero pad) ----------
__global__ __launch_bounds__(256) void conv_w2(const float* __restrict__ W2,
                                               _Float16* __restrict__ w2h) {
  int idx = blockIdx.x * blockDim.x + threadIdx.x;  // 100*1024
  int k = idx & (KP_ - 1);
  int n = idx >> 10;
  w2h[(size_t)n * KP_ + k] = (k < D1_) ? (_Float16)W2[(size_t)n * D1_ + k] : (_Float16)0.f;
}

// ---------- 3) last-point + LSTM step -> h16 [32768][128] (zero-padded) ----------
__global__ __launch_bounds__(256) void point_lstm(const float* __restrict__ routes,
                                                  const float* __restrict__ weff,
                                                  const float* __restrict__ beff,
                                                  const float* __restrict__ c0,
                                                  _Float16* __restrict__ h16) {
  size_t idx = (size_t)blockIdx.x * blockDim.x + threadIdx.x;  // 32768*128
  int j = (int)(idx & (KH_ - 1));
  size_t n = idx >> 7;
  if (j >= DL_) { h16[n * KH_ + j] = (_Float16)0.f; return; }
  const float* lp = routes + (n * T_ + (T_ - 1)) * 3;  // last timestep of row n
  float x0 = lp[0], x1 = lp[1], x2 = lp[2];
  float g[4];
#pragma unroll
  for (int gi = 0; gi < 4; ++gi) {
    const float* wr = weff + (size_t)(gi * DL_ + j) * 3;
    g[gi] = wr[0] * x0 + wr[1] * x1 + wr[2] * x2 + beff[gi * DL_ + j];
  }
  float si = sigmoidf_(g[0]);
  float sf = sigmoidf_(g[1]);
  float tg = tanhf(g[2]);
  float so = sigmoidf_(g[3]);
  float cc = sf * c0[j] + si * tg;
  float hh = so * tanhf(cc);
  h16[n * KH_ + j] = (_Float16)hh;
}

// ---------- 4) main GEMM: relu(h @ W1^T + b1), pooled over M=64 -> P16 ----------
// grid (512 routes, 16 col-tiles of 64); block 512 = 16 waves (4 row x 4 col subtiles)
__global__ __launch_bounds__(512) void gemm_relu_pool(const _Float16* __restrict__ h16,
                                                      const _Float16* __restrict__ w1h,
                                                      const float* __restrict__ b1,
                                                      _Float16* __restrict__ p16) {
  const int r = blockIdx.x;
  const int ct = blockIdx.y;
  const int tid = threadIdx.x;
  const int lane = tid & 31;
  const int w = tid >> 5;     // 0..15
  const int wm = w & 3;       // row subtile
  const int wn = w >> 2;      // col subtile
  const int l16 = lane & 15;
  const int khalf = lane >> 4;

  const int row = r * M_ + wm * 16 + l16;       // < 32768
  const int col = ct * 64 + wn * 16 + l16;      // may be >= 1000 (pad tile)
  const _Float16* arow = h16 + (size_t)row * KH_;

  v8f c = {};
  if (col < D1_) {
    const _Float16* brow = w1h + (size_t)col * KH_;
#pragma unroll
    for (int k = 0; k < KH_; k += 32)
      c = wmma_step(load_a16(arow + k, khalf), load_b16(brow + k, khalf), c);
  } else {
    v16h z = zero16();
#pragma unroll
    for (int k = 0; k < KH_; k += 32)
      c = wmma_step(load_a16(arow + k, khalf), z, c);
  }

  // bias + relu + partial row-sum (C/D layout: lane -> column, VGPR -> row)
  float bias = (col < D1_) ? b1[col] : 0.f;
  float s = 0.f;
#pragma unroll
  for (int v = 0; v < 8; ++v) s += fmaxf(c[v] + bias, 0.f);

  __shared__ float red[16][32];
  red[w][lane] = s;
  __syncthreads();

  if (tid < 64) {  // 4 col-subtiles x 16 columns
    int wn2 = tid >> 4, n = tid & 15;
    float tot = 0.f;
#pragma unroll
    for (int wm2 = 0; wm2 < 4; ++wm2) {
      int wi = wn2 * 4 + wm2;
      tot += red[wi][n] + red[wi][n + 16];  // both row halves
    }
    int gcol = ct * 64 + wn2 * 16 + n;      // 0..1023, covers full padded row
    p16[(size_t)r * KP_ + gcol] = (gcol < D1_) ? (_Float16)tot : (_Float16)0.f;
  }
}

// ---------- 5) emb = P @ W2^T + 64*b2  ([512][100], K=1024) ----------
// grid 32 row-tiles; block 224 = 7 col-tile waves (16*7 = 112 >= 100)
__global__ __launch_bounds__(224) void gemm_emb(const _Float16* __restrict__ p16,
                                                const _Float16* __restrict__ w2h,
                                                const float* __restrict__ b2,
                                                float* __restrict__ emb) {
  const int mt = blockIdx.x;
  const int tid = threadIdx.x;
  const int lane = tid & 31;
  const int w = tid >> 5;  // col tile 0..6
  const int l16 = lane & 15;
  const int khalf = lane >> 4;

  const int row = mt * 16 + l16;   // < 512
  const int col = w * 16 + l16;    // may be >= 100
  const _Float16* arow = p16 + (size_t)row * KP_;

  v8f c = {};
  if (col < DE_) {
    const _Float16* brow = w2h + (size_t)col * KP_;
    for (int k = 0; k < KP_; k += 32)
      c = wmma_step(load_a16(arow + k, khalf), load_b16(brow + k, khalf), c);
  } else {
    v16h z = zero16();
    for (int k = 0; k < KP_; k += 32)
      c = wmma_step(load_a16(arow + k, khalf), z, c);
  }

  int n = lane & 15, mh = lane >> 4;
  int gcol = w * 16 + n;
  if (gcol < DE_) {
    float bb = 64.f * b2[gcol];
#pragma unroll
    for (int v = 0; v < 8; ++v) {
      int grow = mt * 16 + v + 8 * mh;
      emb[(size_t)grow * DE_ + gcol] = c[v] + bb;
    }
  }
}

// ---------- 6) row L2-normalize -> embn16 [512][128] (zero-padded) ----------
__global__ __launch_bounds__(128) void row_normalize(const float* __restrict__ emb,
                                                     _Float16* __restrict__ embn) {
  const int r = blockIdx.x;
  const int t = threadIdx.x;
  __shared__ float red[128];
  float v = (t < DE_) ? emb[(size_t)r * DE_ + t] : 0.f;
  red[t] = v * v;
  __syncthreads();
  for (int s = 64; s > 0; s >>= 1) {
    if (t < s) red[t] += red[t + s];
    __syncthreads();
  }
  float rn = rsqrtf(red[0]);
  embn[(size_t)r * KH_ + t] = (t < DE_) ? (_Float16)(v * rn) : (_Float16)0.f;
}

// ---------- 7) cov = embn @ embn^T, diag -> -inf ([512][512], K=128) ----------
// grid (32,32) tiles, 1 wave per tile
__global__ __launch_bounds__(32) void cov_wmma(const _Float16* __restrict__ embn,
                                               float* __restrict__ out) {
  const int it = blockIdx.x, jt = blockIdx.y;
  const int lane = threadIdx.x;
  const int l16 = lane & 15;
  const int khalf = lane >> 4;

  const _Float16* arow = embn + (size_t)(it * 16 + l16) * KH_;
  const _Float16* brow = embn + (size_t)(jt * 16 + l16) * KH_;

  v8f c = {};
#pragma unroll
  for (int k = 0; k < KH_; k += 32)
    c = wmma_step(load_a16(arow + k, khalf), load_b16(brow + k, khalf), c);

  int n = lane & 15, mh = lane >> 4;
  int j = jt * 16 + n;
#pragma unroll
  for (int v = 0; v < 8; ++v) {
    int i = it * 16 + v + 8 * mh;
    out[(size_t)i * R_ + j] = (i == j) ? -INFINITY : c[v];
  }
}

extern "C" void kernel_launch(void* const* d_in, const int* in_sizes, int n_in,
                              void* d_out, int out_size, void* d_ws, size_t ws_size,
                              hipStream_t stream) {
  const float* routes = (const float*)d_in[0];
  const float* W0  = (const float*)d_in[1];
  const float* b0  = (const float*)d_in[2];
  const float* Wih = (const float*)d_in[3];
  const float* bih = (const float*)d_in[4];
  const float* Whh = (const float*)d_in[5];
  const float* bhh = (const float*)d_in[6];
  const float* W1  = (const float*)d_in[7];
  const float* b1  = (const float*)d_in[8];
  const float* W2  = (const float*)d_in[9];
  const float* b2  = (const float*)d_in[10];
  const float* h0  = (const float*)d_in[11];
  const float* c0  = (const float*)d_in[12];
  float* out = (float*)d_out;

  char* ws = (char*)d_ws;
  float*    weff = (float*)(ws + WEFF_OFF);
  float*    beff = (float*)(ws + BEFF_OFF);
  _Float16* h16  = (_Float16*)(ws + H16_OFF);
  _Float16* w1h  = (_Float16*)(ws + W1H_OFF);
  _Float16* p16  = (_Float16*)(ws + P16_OFF);
  float*    emb  = (float*)(ws + EMB_OFF);
  _Float16* embn = (_Float16*)(ws + EMBN_OFF);
  _Float16* w2h  = (_Float16*)(ws + W2H_OFF);

  prep_eff<<<2, 256, 0, stream>>>(W0, b0, Wih, bih, Whh, bhh, h0, weff, beff);
  conv_w1<<<(D1_ * KH_) / 256, 256, 0, stream>>>(W1, w1h);
  conv_w2<<<(DE_ * KP_) / 256, 256, 0, stream>>>(W2, w2h);
  point_lstm<<<(R_ * M_ * KH_) / 256, 256, 0, stream>>>(routes, weff, beff, c0, h16);
  gemm_relu_pool<<<dim3(R_, 16), 512, 0, stream>>>(h16, w1h, b1, p16);
  gemm_emb<<<32, 224, 0, stream>>>(p16, w2h, b2, emb);
  row_normalize<<<R_, 128, 0, stream>>>(emb, embn);
  cov_wmma<<<dim3(32, 32), 32, 0, stream>>>(embn, out);
}
